// FuncGNN_76510547411041
// MI455X (gfx1250) — matrix operand order repaired and checked by reading, not compiled, not run on hardware
//
#include <hip/hip_runtime.h>
#include <hip/hip_bf16.h>

typedef __bf16 bf16_t;
typedef __attribute__((ext_vector_type(16))) __bf16 v16bf;
typedef __attribute__((ext_vector_type(8)))  __bf16 v8bf;
typedef __attribute__((ext_vector_type(8)))  float  v8f;
typedef __attribute__((ext_vector_type(4)))  int    v4i;

#define NN    50000
#define EE    800000
#define BB    64
#define KK    8
#define FEATD 64
#define EDGED 16
#define HIDD  128
#define TEDD  64
#define NTT   500
#define NCC   3

#if __has_builtin(__builtin_amdgcn_global_load_async_to_lds_b128) && \
    __has_builtin(__builtin_amdgcn_s_wait_asynccnt)
#define USE_ASYNC 1
#else
#define USE_ASYNC 0
#endif

#define AS1 __attribute__((address_space(1)))
#define AS3 __attribute__((address_space(3)))

// ---------------------------------------------------------------- device utils
__device__ __forceinline__ float siluf(float x) { return x / (1.f + __expf(-x)); }
__device__ __forceinline__ float reluf(float x) { return x > 0.f ? x : 0.f; }

__device__ __forceinline__ v8f wmma_bf16(v16bf a, v16bf b, v8f c) {
  return __builtin_amdgcn_wmma_f32_16x16x32_bf16(false, a, false, b,
                                                 (short)0, c, false, false);
}

// async copy of 16 bytes memory -> LDS (per lane), ASYNCcnt-tracked
__device__ __forceinline__ void async_b128(const void* g, void* l) {
#if USE_ASYNC
  __builtin_amdgcn_global_load_async_to_lds_b128(
      (AS1 v4i*)(AS1 void*)g, (AS3 v4i*)(AS3 void*)l, 0, 0);
#else
  *(v8bf*)l = *(const v8bf*)g;
#endif
}
__device__ __forceinline__ void async_wait() {
#if USE_ASYNC
  __builtin_amdgcn_s_wait_asynccnt(0);
#endif
}

// A fragment (16x32 bf16) from an LDS tile, row-major with `stride` halfs/row.
// ISA layout: lanes 0-15 hold M=lane, K=(0..7,16..23); lanes 16-31 K=(8..15,24..31).
__device__ __forceinline__ v16bf load_frag_A(const bf16_t* tile, int stride,
                                             int kt, int lane) {
  int m  = lane & 15;
  int k0 = kt * 32 + ((lane < 16) ? 0 : 8);
  const v8bf* p0 = (const v8bf*)(tile + m * stride + k0);
  const v8bf* p1 = (const v8bf*)(tile + m * stride + k0 + 16);
  v16bf a;
#pragma unroll
  for (int i = 0; i < 8; ++i) { a[i] = (*p0)[i]; a[i + 8] = (*p1)[i]; }
  return a;
}

// B fragment (32x16 bf16) from transposed weights Wt[N][K] (bf16, K mult of 32).
// Lanes 0-15 hold K=(0..7,16..23) of column n, lanes 16-31 hold K=(8..15,24..31).
__device__ __forceinline__ v16bf load_frag_B(const bf16_t* Wt, int K,
                                             int nt, int kt, int lane) {
  int n  = nt * 16 + (lane & 15);
  int k0 = kt * 32 + ((lane < 16) ? 0 : 8);
  const v8bf* p = (const v8bf*)(Wt + (size_t)n * K + k0);
  v16bf b;
#pragma unroll
  for (int i = 0; i < 8; ++i) { b[i] = p[0][i]; b[i + 8] = p[2][i]; }
  return b;
}

// ---------------------------------------------------------------- edge kernel
// One wave = 16 edges. edge1(288->128) silu, edge2(128->128) silu,
// scatter message to agg; optional coord1/coord2 + coord scatter.
__global__ __launch_bounds__(128) void edge_gcl_kernel(
    const bf16_t* __restrict__ hbf, const float* __restrict__ x,
    const int* __restrict__ erow, const int* __restrict__ ecol,
    const float* __restrict__ eattr,
    const bf16_t* __restrict__ W1t, const float* __restrict__ b1,
    const bf16_t* __restrict__ W2t, const float* __restrict__ b2,
    const bf16_t* __restrict__ Wc1t, const float* __restrict__ bc1,
    const bf16_t* __restrict__ Wc2t,
    float* __restrict__ agg, float* __restrict__ cnum, int hasCoord)
{
  __shared__ __align__(16) bf16_t s_in [4][16][288];
  __shared__ __align__(16) bf16_t s_act[4][16][128];
  __shared__ int   s_idx[4][2][16];
  __shared__ float s_w[4][16];

  const int wave = threadIdx.x >> 5;
  const int lane = threadIdx.x & 31;
  const int e0   = blockIdx.x * 64 + wave * 16;
  bf16_t (*tin)[288]  = s_in[wave];
  bf16_t (*tact)[128] = s_act[wave];

  __builtin_prefetch(W1t, 0, 3);
  __builtin_prefetch(W2t, 0, 3);

  float d0 = 0.f, d1 = 0.f, d2 = 0.f;
  if (lane < 16) {
    int e = e0 + lane;
    int r = erow[e], c = ecol[e];
    s_idx[wave][0][lane] = r;
    s_idx[wave][1][lane] = c;
    d0 = x[r * 3 + 0] - x[c * 3 + 0];
    d1 = x[r * 3 + 1] - x[c * 3 + 1];
    d2 = x[r * 3 + 2] - x[c * 3 + 2];
    float radial = d0 * d0 + d1 * d1 + d2 * d2;
#pragma unroll
    for (int j = 0; j < 16; ++j)
      tin[lane][256 + j] = (bf16_t)eattr[(size_t)e * 16 + j];
    tin[lane][272] = (bf16_t)radial;
#pragma unroll
    for (int k = 273; k < 288; ++k) tin[lane][k] = (bf16_t)0.f;
  }
  __syncthreads();

  // gather h[row] (cols 0..127) and h[col] (cols 128..255): async mem->LDS
  for (int t = lane; t < 512; t += 32) {
    int m  = t >> 5;
    int ck = t & 31;
    int node = (ck < 16) ? s_idx[wave][0][m] : s_idx[wave][1][m];
    async_b128(hbf + (size_t)node * HIDD + (ck & 15) * 8, &tin[m][ck * 8]);
  }
  async_wait();
  __syncthreads();

  // edge1: [16x288] @ [288x128] + b1, silu -> s_act
#pragma unroll
  for (int nt = 0; nt < 8; ++nt) {
    v8f acc = {0, 0, 0, 0, 0, 0, 0, 0};
#pragma unroll
    for (int kt = 0; kt < 9; ++kt)
      acc = wmma_bf16(load_frag_A(&tin[0][0], 288, kt, lane),
                      load_frag_B(W1t, 288, nt, kt, lane), acc);
    int n = nt * 16 + (lane & 15);
    float bia = b1[n];
#pragma unroll
    for (int r = 0; r < 8; ++r) {
      int m = r + ((lane >= 16) ? 8 : 0);
      tact[m][n] = (bf16_t)siluf(acc[r] + bia);
    }
  }
  __syncthreads();

  // edge2: + b2, silu -> message m; scatter-add to agg; stash m in tin cols 0..127
#pragma unroll
  for (int nt = 0; nt < 8; ++nt) {
    v8f acc = {0, 0, 0, 0, 0, 0, 0, 0};
#pragma unroll
    for (int kt = 0; kt < 4; ++kt)
      acc = wmma_bf16(load_frag_A(&tact[0][0], 128, kt, lane),
                      load_frag_B(W2t, 128, nt, kt, lane), acc);
    int n = nt * 16 + (lane & 15);
    float bia = b2[n];
#pragma unroll
    for (int r = 0; r < 8; ++r) {
      int m = r + ((lane >= 16) ? 8 : 0);
      float v = siluf(acc[r] + bia);
      atomicAdd(agg + (size_t)s_idx[wave][0][m] * HIDD + n, v);
      tin[m][n] = (bf16_t)v;
    }
  }
  __syncthreads();

  if (hasCoord) {
    // coord1: silu(m @ Wc1 + bc1) -> s_act
#pragma unroll
    for (int nt = 0; nt < 8; ++nt) {
      v8f acc = {0, 0, 0, 0, 0, 0, 0, 0};
#pragma unroll
      for (int kt = 0; kt < 4; ++kt)
        acc = wmma_bf16(load_frag_A(&tin[0][0], 288, kt, lane),
                        load_frag_B(Wc1t, 128, nt, kt, lane), acc);
      int n = nt * 16 + (lane & 15);
      float bia = bc1[n];
#pragma unroll
      for (int r = 0; r < 8; ++r) {
        int m = r + ((lane >= 16) ? 8 : 0);
        tact[m][n] = (bf16_t)siluf(acc[r] + bia);
      }
    }
    __syncthreads();
    // coord2 as WMMA: q[16x128] @ Wc2t (only column 0 meaningful)
    {
      v8f acc = {0, 0, 0, 0, 0, 0, 0, 0};
#pragma unroll
      for (int kt = 0; kt < 4; ++kt)
        acc = wmma_bf16(load_frag_A(&tact[0][0], 128, kt, lane),
                        load_frag_B(Wc2t, 128, 0, kt, lane), acc);
      if ((lane & 15) == 0) {
        int mo = (lane >= 16) ? 8 : 0;
#pragma unroll
        for (int r = 0; r < 8; ++r) s_w[wave][r + mo] = acc[r];
      }
    }
    __syncthreads();
    if (lane < 16) {  // scatter coord_diff * w
      float w_e = s_w[wave][lane];
      int rn = s_idx[wave][0][lane];
      atomicAdd(cnum + (size_t)rn * 3 + 0, d0 * w_e);
      atomicAdd(cnum + (size_t)rn * 3 + 1, d1 * w_e);
      atomicAdd(cnum + (size_t)rn * 3 + 2, d2 * w_e);
    }
  }
}

// ---------------------------------------------------------------- node kernel
// node1([h,agg,h0]) silu, node2, residual; optional coord mean-update.
__global__ __launch_bounds__(64) void node_gcl_kernel(
    bf16_t* __restrict__ hbf, float* __restrict__ hf32,
    const float* __restrict__ agg, const bf16_t* __restrict__ h0bf, int attrDim,
    const bf16_t* __restrict__ W1t, const float* __restrict__ b1,
    const bf16_t* __restrict__ W2t, const float* __restrict__ b2,
    float* __restrict__ xio, const float* __restrict__ cnum,
    const float* __restrict__ cntN, int hasCoord, int Ntot)
{
  __shared__ __align__(16) bf16_t s_in [2][16][384];
  __shared__ __align__(16) bf16_t s_act[2][16][128];
  const int wave = threadIdx.x >> 5;
  const int lane = threadIdx.x & 31;
  const int n0   = blockIdx.x * 32 + wave * 16;
  const int K    = 256 + attrDim;           // 256 / 320 / 384, all mult of 32
  bf16_t (*tin)[384]  = s_in[wave];
  bf16_t (*tact)[128] = s_act[wave];

  const int cpr = K >> 3;                   // 8-half chunks per row
  for (int t = lane; t < 16 * cpr; t += 32) {
    int m  = t / cpr;
    int kb = (t % cpr) * 8;
    int node = n0 + m; if (node >= Ntot) node = Ntot - 1;
    if (kb < 128) {
      async_b128(hbf + (size_t)node * HIDD + kb, &tin[m][kb]);
    } else if (kb < 256) {
      const float* s = agg + (size_t)node * HIDD + (kb - 128);
      v8bf v;
#pragma unroll
      for (int i = 0; i < 8; ++i) v[i] = (bf16_t)s[i];
      *(v8bf*)(&tin[m][kb]) = v;
    } else {
      async_b128(h0bf + (size_t)node * attrDim + (kb - 256), &tin[m][kb]);
    }
  }
  async_wait();
  __syncthreads();

  const int nkt = K >> 5;
  for (int nt = 0; nt < 8; ++nt) {
    v8f acc = {0, 0, 0, 0, 0, 0, 0, 0};
    for (int kt = 0; kt < nkt; ++kt)
      acc = wmma_bf16(load_frag_A(&tin[0][0], 384, kt, lane),
                      load_frag_B(W1t, K, nt, kt, lane), acc);
    int n = nt * 16 + (lane & 15);
    float bia = b1[n];
#pragma unroll
    for (int r = 0; r < 8; ++r) {
      int m = r + ((lane >= 16) ? 8 : 0);
      tact[m][n] = (bf16_t)siluf(acc[r] + bia);
    }
  }
  __syncthreads();

#pragma unroll
  for (int nt = 0; nt < 8; ++nt) {
    v8f acc = {0, 0, 0, 0, 0, 0, 0, 0};
#pragma unroll
    for (int kt = 0; kt < 4; ++kt)
      acc = wmma_bf16(load_frag_A(&tact[0][0], 128, kt, lane),
                      load_frag_B(W2t, 128, nt, kt, lane), acc);
    int n = nt * 16 + (lane & 15);
    float bia = b2[n];
#pragma unroll
    for (int r = 0; r < 8; ++r) {
      int m = r + ((lane >= 16) ? 8 : 0);
      int node = n0 + m;
      if (node < Ntot) {
        float h = hf32[(size_t)node * HIDD + n] + acc[r] + bia;  // recurrent
        hf32[(size_t)node * HIDD + n] = h;
        hbf [(size_t)node * HIDD + n] = (bf16_t)h;
      }
    }
  }

  if (hasCoord && lane < 16) {
    int node = n0 + lane;
    if (node < Ntot) {
      float c = cntN[node]; c = c > 1.f ? c : 1.f;
#pragma unroll
      for (int d = 0; d < 3; ++d)
        xio[(size_t)node * 3 + d] += cnum[(size_t)node * 3 + d] / c;
    }
  }
}

// ---------------------------------------------------------------- generic row GEMM
// out[M,Nout] = act(A[M,K] @ Wt^T + bias). ACT: 0=none, 1=silu, 2=relu.
template <int ACT>
__global__ __launch_bounds__(64) void rowgemm_kernel(
    const bf16_t* __restrict__ A, int M, int K,
    const bf16_t* __restrict__ Wt, const float* __restrict__ bias, int Nout,
    float* __restrict__ outF, bf16_t* __restrict__ outB)
{
  extern __shared__ bf16_t sm[];
  const int wave = threadIdx.x >> 5;
  const int lane = threadIdx.x & 31;
  bf16_t* tin = sm + (size_t)wave * 16 * K;
  const int r0  = blockIdx.x * 32 + wave * 16;
  const int cpr = K >> 3;

  __builtin_prefetch(Wt, 0, 3);
  for (int t = lane; t < 16 * cpr; t += 32) {
    int m  = t / cpr;
    int kb = (t % cpr) * 8;
    int row = r0 + m; if (row >= M) row = M - 1;
    async_b128(A + (size_t)row * K + kb, tin + m * K + kb);
  }
  async_wait();
  __syncthreads();

  const int nkt = K >> 5, nnt = Nout >> 4;
  for (int nt = 0; nt < nnt; ++nt) {
    v8f acc = {0, 0, 0, 0, 0, 0, 0, 0};
    for (int kt = 0; kt < nkt; ++kt)
      acc = wmma_bf16(load_frag_A(tin, K, kt, lane),
                      load_frag_B(Wt, K, nt, kt, lane), acc);
    int n = nt * 16 + (lane & 15);
    float bia = bias[n];
#pragma unroll
    for (int r = 0; r < 8; ++r) {
      int m = r + ((lane >= 16) ? 8 : 0);
      int row = r0 + m;
      if (row < M) {
        float v = acc[r] + bia;
        if (ACT == 1) v = siluf(v); else if (ACT == 2) v = reluf(v);
        if (outF) outF[(size_t)row * Nout + n] = v;
        if (outB) outB[(size_t)row * Nout + n] = (bf16_t)v;
      }
    }
  }
}

// ---------------------------------------------------------------- small kernels
__global__ void convW_kernel(const float* __restrict__ W, bf16_t* __restrict__ Wt,
                             int Ksrc, int Nsrc, int Kpad, int Npad, int mode) {
  int i = blockIdx.x * blockDim.x + threadIdx.x;
  if (i >= Npad * Kpad) return;
  int n = i / Kpad, k = i % Kpad;
  int ks;
  if (mode == 1)  // edge1 column permutation: [h|h|eattr(16)|radial|pad]
    ks = (k < 256) ? k : (k < 272 ? k + 1 : (k == 272 ? 256 : -1));
  else
    ks = (k < Ksrc) ? k : -1;
  float v = (ks >= 0 && n < Nsrc) ? W[(size_t)ks * Nsrc + n] : 0.f;
  Wt[(size_t)n * Kpad + k] = (bf16_t)v;
}

__global__ void f32_to_bf16_kernel(const float* __restrict__ s,
                                   bf16_t* __restrict__ d, int n) {
  int i = blockIdx.x * blockDim.x + threadIdx.x;
  if (i < n) d[i] = (bf16_t)s[i];
}

__global__ void count_rows_kernel(const int* __restrict__ erow, float* cntN) {
  int i = blockIdx.x * blockDim.x + threadIdx.x;
  if (i < EE) atomicAdd(cntN + erow[i], 1.f);
}

__global__ void pool_cnt_kernel(const int* __restrict__ batch, float* cntB) {
  int i = blockIdx.x * blockDim.x + threadIdx.x;
  if (i < NN) atomicAdd(cntB + batch[i], 1.f);
}

__global__ void pool_sum_kernel(const float* __restrict__ hf,
                                const int* __restrict__ batch, float* psum) {
  int i = blockIdx.x * blockDim.x + threadIdx.x;
  if (i < NN * HIDD)
    atomicAdd(psum + (size_t)batch[i / HIDD] * HIDD + (i % HIDD), hf[i]);
}

__global__ void build_z0_kernel(const float* __restrict__ psum,
                                const float* __restrict__ cntB,
                                const int* __restrict__ tasks,
                                const float* __restrict__ te,
                                bf16_t* __restrict__ z0) {
  int i = blockIdx.x * blockDim.x + threadIdx.x;
  if (i >= BB * KK * (HIDD + TEDD)) return;
  int r = i / (HIDD + TEDD), c = i % (HIDD + TEDD);
  float v;
  if (c < HIDD) {
    int b = r / KK;
    float cnt = cntB[b]; cnt = cnt > 1.f ? cnt : 1.f;
    v = psum[(size_t)b * HIDD + c] / cnt;
  } else {
    int t = tasks[r * 2 + 1];
    v = te[(size_t)t * TEDD + (c - HIDD)];
  }
  z0[(size_t)r * (HIDD + TEDD) + c] = (bf16_t)v;
}

__global__ void pad_bias_kernel(const float* __restrict__ b, float* dst) {
  int i = blockIdx.x * blockDim.x + threadIdx.x;
  if (i < 16) dst[i] = (i < NCC) ? b[i] : 0.f;
}

__global__ void extract_out_kernel(const float* __restrict__ t,
                                   float* __restrict__ out) {
  int i = blockIdx.x * blockDim.x + threadIdx.x;
  if (i < BB * KK * NCC) out[i] = t[(size_t)(i / NCC) * 16 + (i % NCC)];
}

// ---------------------------------------------------------------- host launcher
extern "C" void kernel_launch(void* const* d_in, const int* in_sizes, int n_in,
                              void* d_out, int out_size, void* d_ws, size_t ws_size,
                              hipStream_t stream) {
  (void)in_sizes; (void)n_in; (void)out_size; (void)ws_size;

  const float* h_in  = (const float*)d_in[0];
  const float* x_in  = (const float*)d_in[1];
  const int*   erow  = (const int*)d_in[2];
  const int*   ecol  = erow + EE;
  const float* eatt  = (const float*)d_in[3];
  const int*   batch = (const int*)d_in[4];
  const int*   tasks = (const int*)d_in[5];
  auto F = [&](int i) { return (const float*)d_in[i]; };

  // ---- workspace bump allocator
  char* base = (char*)d_ws; size_t off = 0;
  auto alloc = [&](size_t bytes) -> char* {
    char* p = base + off;
    off = (off + bytes + 255) & ~(size_t)255;
    return p;
  };
  float*  xw    = (float*) alloc((size_t)NN * 3 * 4);
  float*  hF    = (float*) alloc((size_t)NN * HIDD * 4);
  bf16_t* hB    = (bf16_t*)alloc((size_t)NN * HIDD * 2);
  bf16_t* h0bf  = (bf16_t*)alloc((size_t)NN * HIDD * 2);
  float*  agg   = (float*) alloc((size_t)NN * HIDD * 4);
  float*  cnum  = (float*) alloc((size_t)NN * 3 * 4);
  float*  cntN  = (float*) alloc((size_t)NN * 4);
  float*  psum  = (float*) alloc((size_t)BB * HIDD * 4);
  float*  cntB  = (float*) alloc((size_t)BB * 4);
  bf16_t* z0    = (bf16_t*)alloc((size_t)512 * 192 * 2);
  bf16_t* t1    = (bf16_t*)alloc((size_t)512 * 128 * 2);
  bf16_t* t2    = (bf16_t*)alloc((size_t)512 * 128 * 2);
  bf16_t* t3    = (bf16_t*)alloc((size_t)512 * 128 * 2);
  float*  outT  = (float*) alloc((size_t)512 * 16 * 4);
  float*  b3pad = (float*) alloc(16 * 4);

  auto cvt = [&](int wIdx, bf16_t* Wt, int Ksrc, int Nsrc, int Kpad, int Npad,
                 int mode) {
    int total = Npad * Kpad;
    convW_kernel<<<(total + 255) / 256, 256, 0, stream>>>(
        F(wIdx), Wt, Ksrc, Nsrc, Kpad, Npad, mode);
  };

  // ---- parameter leaf indices (jax pytree: sorted dict keys, list order)
  // params: egnns(6..97) < mlp(98..105) < pool_gcl(106..113) < task_embed(114)
  struct GclW {
    bf16_t *e1, *e2, *c1, *c2, *n1, *n2;
    const float *b1, *b2, *bc1, *nb1, *nb2;
    int n1K;
  };
  GclW gw[8];
  bf16_t* emb[2];
  const float* embB[2];
  for (int eg = 0; eg < 2; ++eg) {
    int ebase = (eg == 0) ? 6 : 52;
    embB[eg] = F(ebase + 0);
    int embK = (eg == 0) ? FEATD : HIDD;
    emb[eg]  = (bf16_t*)alloc((size_t)HIDD * embK * 2);
    cvt(ebase + 1, emb[eg], embK, HIDD, embK, HIDD, 0);
    int n1K = (eg == 0) ? 320 : 384;
    for (int j = 0; j < 4; ++j) {
      int gb = ebase + 2 + 11 * j;  // coord1.b,w coord2.w edge1.b,w edge2.b,w node1.b,w node2.b,w
      GclW& g = gw[eg * 4 + j];
      g.e1 = (bf16_t*)alloc((size_t)HIDD * 288 * 2);
      g.e2 = (bf16_t*)alloc((size_t)HIDD * 128 * 2);
      g.c1 = (bf16_t*)alloc((size_t)HIDD * 128 * 2);
      g.c2 = (bf16_t*)alloc((size_t)16 * 128 * 2);
      g.n1 = (bf16_t*)alloc((size_t)HIDD * n1K * 2);
      g.n2 = (bf16_t*)alloc((size_t)HIDD * 128 * 2);
      g.n1K = n1K;
      g.bc1 = F(gb + 0); cvt(gb + 1, g.c1, 128, 128, 128, 128, 0);
      cvt(gb + 2, g.c2, 128, 1, 128, 16, 0);  // coord2.w [128,1] -> B col 0
      g.b1  = F(gb + 3); cvt(gb + 4, g.e1, 273, 128, 288, 128, 1);
      g.b2  = F(gb + 5); cvt(gb + 6, g.e2, 128, 128, 128, 128, 0);
      g.nb1 = F(gb + 7); cvt(gb + 8, g.n1, n1K, 128, n1K, 128, 0);
      g.nb2 = F(gb + 9); cvt(gb + 10, g.n2, 128, 128, 128, 128, 0);
    }
  }
  // mlp
  bf16_t* mw0 = (bf16_t*)alloc((size_t)128 * 192 * 2);
  bf16_t* mw1 = (bf16_t*)alloc((size_t)128 * 128 * 2);
  bf16_t* mw2 = (bf16_t*)alloc((size_t)128 * 128 * 2);
  bf16_t* mw3 = (bf16_t*)alloc((size_t)16 * 128 * 2);
  cvt(99, mw0, 192, 128, 192, 128, 0);
  cvt(101, mw1, 128, 128, 128, 128, 0);
  cvt(103, mw2, 128, 128, 128, 128, 0);
  cvt(105, mw3, 128, NCC, 128, 16, 0);
  pad_bias_kernel<<<1, 16, 0, stream>>>(F(104), b3pad);
  // pool gcl
  bf16_t* pe1 = (bf16_t*)alloc((size_t)HIDD * 288 * 2);
  bf16_t* pe2 = (bf16_t*)alloc((size_t)HIDD * 128 * 2);
  bf16_t* pn1 = (bf16_t*)alloc((size_t)HIDD * 256 * 2);
  bf16_t* pn2 = (bf16_t*)alloc((size_t)HIDD * 128 * 2);
  cvt(107, pe1, 273, 128, 288, 128, 1);
  cvt(109, pe2, 128, 128, 128, 128, 0);
  cvt(111, pn1, 256, 128, 256, 128, 0);
  cvt(113, pn2, 128, 128, 128, 128, 0);
  const float* taskE = F(114);

  // ---- pipeline
  (void)hipMemcpyAsync(xw, x_in, (size_t)NN * 3 * 4, hipMemcpyDeviceToDevice, stream);
  (void)hipMemsetAsync(cntN, 0, (size_t)NN * 4, stream);
  count_rows_kernel<<<(EE + 255) / 256, 256, 0, stream>>>(erow, cntN);

  const int nodeGrid = (NN + 31) / 32;
  for (int eg = 0; eg < 2; ++eg) {
    int attr = (eg == 0) ? FEATD : HIDD;
    if (eg == 0) {
      f32_to_bf16_kernel<<<(NN * FEATD + 255) / 256, 256, 0, stream>>>(
          h_in, h0bf, NN * FEATD);
      rowgemm_kernel<0><<<nodeGrid, 64, 2 * 16 * FEATD * 2, stream>>>(
          h0bf, NN, FEATD, emb[0], embB[0], 128, hF, hB);
    } else {
      (void)hipMemcpyAsync(h0bf, hB, (size_t)NN * HIDD * 2,
                           hipMemcpyDeviceToDevice, stream);
      rowgemm_kernel<0><<<nodeGrid, 64, 2 * 16 * HIDD * 2, stream>>>(
          h0bf, NN, HIDD, emb[1], embB[1], 128, hF, hB);
    }
    for (int j = 0; j < 4; ++j) {
      GclW& g = gw[eg * 4 + j];
      (void)hipMemsetAsync(agg, 0, (size_t)NN * HIDD * 4, stream);
      (void)hipMemsetAsync(cnum, 0, (size_t)NN * 3 * 4, stream);
      edge_gcl_kernel<<<EE / 64, 128, 0, stream>>>(
          hB, xw, erow, ecol, eatt, g.e1, g.b1, g.e2, g.b2, g.c1, g.bc1,
          g.c2, agg, cnum, 1);
      node_gcl_kernel<<<nodeGrid, 64, 0, stream>>>(
          hB, hF, agg, h0bf, attr, g.n1, g.nb1, g.n2, g.nb2,
          xw, cnum, cntN, 1, NN);
    }
  }

  // pool GCL (no coord update, no node attr)
  (void)hipMemsetAsync(agg, 0, (size_t)NN * HIDD * 4, stream);
  edge_gcl_kernel<<<EE / 64, 128, 0, stream>>>(
      hB, xw, erow, ecol, eatt, pe1, F(106), pe2, F(108), pe2, F(108),
      pe2, agg, cnum, 0);
  node_gcl_kernel<<<nodeGrid, 64, 0, stream>>>(
      hB, hF, agg, h0bf, 0, pn1, F(110), pn2, F(112), xw, cnum, cntN, 0, NN);

  // mean pool over graphs
  (void)hipMemsetAsync(psum, 0, (size_t)BB * HIDD * 4, stream);
  (void)hipMemsetAsync(cntB, 0, (size_t)BB * 4, stream);
  pool_cnt_kernel<<<(NN + 255) / 256, 256, 0, stream>>>(batch, cntB);
  pool_sum_kernel<<<(NN * HIDD + 255) / 256, 256, 0, stream>>>(hF, batch, psum);

  // task head
  build_z0_kernel<<<(512 * 192 + 255) / 256, 256, 0, stream>>>(
      psum, cntB, tasks, taskE, z0);
  rowgemm_kernel<2><<<16, 64, 2 * 16 * 192 * 2, stream>>>(
      z0, 512, 192, mw0, F(98), 128, nullptr, t1);
  rowgemm_kernel<2><<<16, 64, 2 * 16 * 128 * 2, stream>>>(
      t1, 512, 128, mw1, F(100), 128, nullptr, t2);
  rowgemm_kernel<2><<<16, 64, 2 * 16 * 128 * 2, stream>>>(
      t2, 512, 128, mw2, F(102), 128, nullptr, t3);
  rowgemm_kernel<0><<<16, 64, 2 * 16 * 128 * 2, stream>>>(
      t3, 512, 128, mw3, b3pad, 16, outT, nullptr);
  extract_out_kernel<<<(512 * NCC + 255) / 256, 256, 0, stream>>>(
      outT, (float*)d_out);
}
